// SentenceTokenTypeMoeBlock_44667659878787
// MI455X (gfx1250) — compile-verified, matrix operand order checked by reading, not run on hardware
//
#include <hip/hip_runtime.h>

// ---------------------------------------------------------------------------
// Problem constants (match reference)
// ---------------------------------------------------------------------------
#define HID   768
#define FFD   3072
#define NE    8
#define BT    16384            // B*T tokens

#define TILE_M  48             // tokens per block (3 WMMA M sub-tiles)
#define MTS     3              // M sub-tiles
#define NTILES  ((BT + TILE_M - 1) / TILE_M)   // 342
#define TPB     512            // 16 waves per block
#define WAVES   16

#define KC1     (HID / 32)     // 24  K-chunks for GEMM1
#define NT1     (FFD / 16)     // 192 N-tiles  for GEMM1
#define KC2     (FFD / 32)     // 96  K-chunks for GEMM2
#define NT2     (HID / 16)     // 48  N-tiles  for GEMM2
#define FCW     256            // fused FF chunk width (16 n1-tiles, 8 k2-chunks)
#define FCN     (FFD / FCW)    // 12
#define K2C     (FCW / 32)     // 8

#define XPACK_ELEMS (MTS * KC1 * 512)   // 36864 bf16 = 72 KB
#define HPACK_ELEMS (MTS * K2C * 512)   // 12288 bf16 = 24 KB
#define SMEM_BYTES  (XPACK_ELEMS * 2 + HPACK_ELEMS * 2 + TILE_M * 4)

typedef __attribute__((ext_vector_type(16))) __bf16 v16bf;
typedef __attribute__((ext_vector_type(8)))  float  v8f;

// K index inside a 32-wide chunk for WMMA 16-bit A/B lane layout (ISA 7.12.2):
//   vgpr v = s>>1, half h = s&1 ; K = 16*(v>=4) + 8*(lane>=16) + 2*(v&3) + h
__device__ __forceinline__ int kmap(int s, int la) {
  int v = s >> 1, h = s & 1;
  return ((v >= 4) ? 16 : 0) + ((la >= 16) ? 8 : 0) + ((v & 3) << 1) + h;
}

__device__ __forceinline__ float gelu_tanh(float x) {
  float x3 = x * x * x;
  return 0.5f * x * (1.0f + tanhf(0.7978845608028654f * (x + 0.044715f * x3)));
}

__device__ __forceinline__ v8f wmma_bf16(v16bf a, v16bf b, v8f c) {
  return __builtin_amdgcn_wmma_f32_16x16x32_bf16(false, a, false, b, (short)0, c,
                                                 false, false);
}

// WGP-scope prefetch (scope field 0): pulls the line into ALL cache levels,
// so the later global_load_b128 is a near-cache hit. (__builtin_prefetch only
// produces scope:SCOPE_SE, which stops at GL2 — useless for L2-resident
// weights.) Addresses must be valid: WGP-scope prefetch is non-speculative.
__device__ __forceinline__ void prefetch_wgp(const void* p) {
  asm volatile("global_prefetch_b8 %0, off" :: "v"(p));
}

// ---------------------------------------------------------------------------
// Kernel 1: zero per-expert counters
// ---------------------------------------------------------------------------
__global__ void zero_counts_kernel(int* counts) {
  if (threadIdx.x < NE) counts[threadIdx.x] = 0;
}

// ---------------------------------------------------------------------------
// Kernel 2: route tokens -> per-expert index lists (fixed BT-sized regions)
// ---------------------------------------------------------------------------
__global__ void route_kernel(const int* __restrict__ tt, int* __restrict__ counts,
                             int* __restrict__ perm) {
  int t = blockIdx.x * blockDim.x + threadIdx.x;
  if (t < BT) {
    int e = tt[t];
    int slot = atomicAdd(&counts[e], 1);
    perm[e * BT + slot] = t;
  }
}

// ---------------------------------------------------------------------------
// Kernel 3: pack f32 weights [NE][K][N] -> bf16 WMMA B-operand tiles
//           dst layout: [e][n_tile][k_chunk][lane(32)][slot(16)]
// ---------------------------------------------------------------------------
__global__ __launch_bounds__(256) void pack_w_kernel(const float* __restrict__ src,
                                                     __bf16* __restrict__ dst,
                                                     int Kdim, int Ndim, int NT, int KC) {
  long i = (long)blockIdx.x * 256 + threadIdx.x;
  long total = (long)NE * NT * KC * 512;
  if (i >= total) return;
  int s  = (int)(i & 15);
  int la = (int)((i >> 4) & 31);
  long rest = i >> 9;
  int kc = (int)(rest % KC); rest /= KC;
  int n  = (int)(rest % NT);
  int e  = (int)(rest / NT);
  int Kg  = kc * 32 + kmap(s, la);
  int col = n * 16 + (la & 15);
  dst[i] = (__bf16)src[((size_t)e * Kdim + Kg) * Ndim + col];
}

// ---------------------------------------------------------------------------
// Kernel 4: grouped fused MoE MLP.
//   One block = (expert e, tile of 48 tokens). 512 threads = 16 waves.
//   Per 256-wide FF chunk: wave w does GEMM1 n1-tile (fc*16+w) x 3 M sub-tiles;
//   GEMM2 accumulators: n2-tiles w*3 .. w*3+2 x 3 M sub-tiles (72 VGPRs).
//   Latency hiding: WGP-scope global_prefetch (no VGPR cost) + batched loads.
// ---------------------------------------------------------------------------
__global__ __launch_bounds__(TPB) void moe_kernel(
    const float* __restrict__ x, const float* __restrict__ b1,
    const float* __restrict__ b2, const __bf16* __restrict__ w1t,
    const __bf16* __restrict__ w2t, const int* __restrict__ counts,
    const int* __restrict__ perm, float* __restrict__ out) {
  int e    = blockIdx.x / NTILES;
  int tile = blockIdx.x % NTILES;
  int cnt  = counts[e];
  int start = tile * TILE_M;
  if (start >= cnt) return;   // block-uniform: EXEC stays all-ones for WMMA

  extern __shared__ __bf16 smem[];
  __bf16* hpack = smem;                      // [mt][k2][lane][slot]  (low: DS imm offsets)
  __bf16* xpack = smem + HPACK_ELEMS;        // [mt][kc][lane][slot]
  int*    toks  = (int*)(smem + HPACK_ELEMS + XPACK_ELEMS);

  int tid  = threadIdx.x;
  int lane = tid & 31;
  int wave = tid >> 5;
  int hi   = lane >> 4;

  // Prefetch fc=0 GEMM1 B tiles while we stage x.
  {
    const __bf16* wb1f = w1t + ((size_t)(e * NT1 + wave) * KC1) * 512 + lane * 16;
#pragma unroll
    for (int t = 0; t < 4; ++t)
      prefetch_wgp(wb1f + (size_t)t * 512);
  }

  if (tid < TILE_M) {
    int sidx = start + tid;
    toks[tid] = (sidx < cnt) ? perm[e * BT + sidx] : 0;  // pad rows use token 0, never stored
  }
  __syncthreads();

  // Stage x tile (48 x 768) into WMMA A-operand lane layout.
  for (int i = tid; i < XPACK_ELEMS; i += TPB) {
    int s  = i & 15;
    int la = (i >> 4) & 31;
    int kc = (i >> 9) % KC1;
    int mt = i / (512 * KC1);
    int K  = kmap(s, la);
    int tok = toks[mt * 16 + (la & 15)];
    xpack[i] = (__bf16)x[(size_t)tok * HID + kc * 32 + K];
  }
  __syncthreads();

  // Per-mt LDS base pointers -> ds_load_b128 with immediate offsets.
  const __bf16* xpb[MTS];
  const __bf16* hpb[MTS];
#pragma unroll
  for (int mt = 0; mt < MTS; ++mt) {
    xpb[mt] = xpack + (size_t)(mt * KC1) * 512 + lane * 16;
    hpb[mt] = hpack + (size_t)(mt * K2C) * 512 + lane * 16;
  }

  v8f acc[MTS][3];
#pragma unroll
  for (int mt = 0; mt < MTS; ++mt)
#pragma unroll
    for (int jj = 0; jj < 3; ++jj) acc[mt][jj] = {};

  for (int fc = 0; fc < FCN; ++fc) {
    int n1g = fc * WAVES + wave;
    const __bf16* wb1 = w1t + ((size_t)(e * NT1 + n1g) * KC1) * 512 + lane * 16;
    const __bf16* wb2 = w2t + ((size_t)(e * NT2 + wave * 3) * KC2 + fc * K2C) * 512 +
                        lane * 16;           // tile (jj,k2) at + (jj*KC2 + k2)*512

    // Prefetch this fc's first GEMM2 B tiles (consumed after GEMM1 + repack).
#pragma unroll
    for (int jj = 0; jj < 3; ++jj) {
      prefetch_wgp(wb2 + (size_t)jj * KC2 * 512);
      prefetch_wgp(wb2 + ((size_t)jj * KC2 + 1) * 512);
      prefetch_wgp(wb2 + ((size_t)jj * KC2 + 2) * 512);
    }

    // ---- GEMM1: this wave's n1 tile, 3 M sub-tiles, depth-4 B window -----
    v8f d[MTS];
#pragma unroll
    for (int mt = 0; mt < MTS; ++mt) d[mt] = {};

    v16bf bq[4];
#pragma unroll
    for (int t = 0; t < 4; ++t) bq[t] = *(const v16bf*)(wb1 + (size_t)t * 512);
#pragma unroll
    for (int kc = 0; kc < KC1; ++kc) {
      if (kc + 8 < KC1)
        prefetch_wgp(wb1 + (size_t)(kc + 8) * 512);
#pragma unroll
      for (int mt = 0; mt < MTS; ++mt) {
        v16bf a = *(const v16bf*)(xpb[mt] + (size_t)kc * 512);
        d[mt] = wmma_bf16(a, bq[kc & 3], d[mt]);
      }
      if (kc + 4 < KC1)
        bq[kc & 3] = *(const v16bf*)(wb1 + (size_t)(kc + 4) * 512);
    }

    // Prefetch next fc's first GEMM1 B tiles (consumed after this GEMM2).
    if (fc + 1 < FCN) {
      const __bf16* wb1n = wb1 + (size_t)WAVES * KC1 * 512;
#pragma unroll
      for (int t = 0; t < 4; ++t)
        prefetch_wgp(wb1n + (size_t)t * 512);
    }

    // ---- bias + GELU, repack D tiles as GEMM2 A-operand chunks -----------
    int cw   = wave * 16 + (lane & 15);      // column within 256-wide FF chunk
    float bias = b1[e * FFD + fc * FCW + cw];
    int k2l  = cw >> 5;                      // k2 chunk inside this FF chunk (0..7)
    int Koff = cw & 31;                      // K offset inside chunk
    int rem  = Koff & 15;
    int hi2  = (rem >= 8) ? 1 : 0;
    int v2   = ((Koff >= 16) ? 4 : 0) + ((rem & 7) >> 1);
    int s2   = (v2 << 1) + (Koff & 1);

    __syncthreads();                         // previous chunk's GEMM2 reads done
#pragma unroll
    for (int r = 0; r < 8; ++r) {
      int row = r + 8 * hi;                  // D layout: vgpr r holds M = r + 8*(lane>=16)
#pragma unroll
      for (int mt = 0; mt < MTS; ++mt) {
        float g = gelu_tanh(d[mt][r] + bias);
        hpack[((mt * K2C + k2l) * 32 + (row + 16 * hi2)) * 16 + s2] = (__bf16)g;
      }
    }
    __syncthreads();

    // ---- GEMM2 partial: k2-outer, 3 n2-tiles, prefetch 3 k2-steps ahead --
    v16bf cq[2][3];
#pragma unroll
    for (int jj = 0; jj < 3; ++jj)
      cq[0][jj] = *(const v16bf*)(wb2 + (size_t)jj * KC2 * 512);
#pragma unroll
    for (int k2 = 0; k2 < K2C; ++k2) {
      int cur = k2 & 1, nxt = cur ^ 1;
      if (k2 + 3 < K2C) {
#pragma unroll
        for (int jj = 0; jj < 3; ++jj)
          prefetch_wgp(wb2 + ((size_t)jj * KC2 + (k2 + 3)) * 512);
      }
      if (k2 + 1 < K2C) {
#pragma unroll
        for (int jj = 0; jj < 3; ++jj)
          cq[nxt][jj] = *(const v16bf*)(wb2 + ((size_t)jj * KC2 + (k2 + 1)) * 512);
      }
      v16bf a[MTS];
#pragma unroll
      for (int mt = 0; mt < MTS; ++mt)
        a[mt] = *(const v16bf*)(hpb[mt] + (size_t)k2 * 512);
#pragma unroll
      for (int jj = 0; jj < 3; ++jj)
#pragma unroll
        for (int mt = 0; mt < MTS; ++mt)
          acc[mt][jj] = wmma_bf16(a[mt], cq[cur][jj], acc[mt][jj]);
    }
  }

  // ---- epilogue: + b2, scatter rows back to token positions --------------
#pragma unroll
  for (int mt = 0; mt < MTS; ++mt) {
#pragma unroll
    for (int jj = 0; jj < 3; ++jj) {
      int col = (wave * 3 + jj) * 16 + (lane & 15);
      float b2v = b2[e * HID + col];
#pragma unroll
      for (int r = 0; r < 8; ++r) {
        int rloc = mt * 16 + r + 8 * hi;
        if (start + rloc < cnt) {
          int tok = toks[rloc];
          out[(size_t)tok * HID + col] = acc[mt][jj][r] + b2v;
        }
      }
    }
  }
}

// ---------------------------------------------------------------------------
// Host launcher. ws layout:
//   [0)            counts      : NE ints
//   [256)          perm        : NE*BT ints              (512 KB)
//   [1 MB)         w1t (bf16)  : NE*NT1*KC1*512 elems    (36 MB)
//   [1 MB + 36 MB) w2t (bf16)  : NE*NT2*KC2*512 elems    (36 MB)
// ---------------------------------------------------------------------------
extern "C" void kernel_launch(void* const* d_in, const int* in_sizes, int n_in,
                              void* d_out, int out_size, void* d_ws, size_t ws_size,
                              hipStream_t stream) {
  const float* x  = (const float*)d_in[0];
  const int*   tt = (const int*)d_in[1];
  const float* w1 = (const float*)d_in[2];
  const float* b1 = (const float*)d_in[3];
  const float* w2 = (const float*)d_in[4];
  const float* b2 = (const float*)d_in[5];
  float* out = (float*)d_out;

  char* ws = (char*)d_ws;
  int*    counts = (int*)ws;
  int*    perm   = (int*)(ws + 256);
  __bf16* w1t    = (__bf16*)(ws + (1u << 20));
  __bf16* w2t    = (__bf16*)(ws + (1u << 20) + (size_t)NE * NT1 * KC1 * 512 * 2);

  zero_counts_kernel<<<1, 64, 0, stream>>>(counts);
  route_kernel<<<BT / 256, 256, 0, stream>>>(tt, counts, perm);

  long pk_total = (long)NE * NT1 * KC1 * 512;  // same count for both tensors
  int pk_blocks = (int)((pk_total + 255) / 256);
  pack_w_kernel<<<pk_blocks, 256, 0, stream>>>(w1, w1t, HID, FFD, NT1, KC1);
  pack_w_kernel<<<pk_blocks, 256, 0, stream>>>(w2, w2t, FFD, HID, NT2, KC2);

  moe_kernel<<<NE * NTILES, TPB, SMEM_BYTES, stream>>>(x, b1, b2, w1t, w2t, counts,
                                                       perm, out);
}
